// PointNetFeaturePropagation_21182778704779
// MI455X (gfx1250) — matrix operand order repaired and checked by reading, not compiled, not run on hardware
//
#include <hip/hip_runtime.h>
#include <hip/hip_bf16.h>
#include <stdint.h>

typedef __attribute__((ext_vector_type(16))) __bf16 v16bf;
typedef __attribute__((ext_vector_type(8)))  float  v8f;

static __device__ __forceinline__ unsigned short f2bf(float f) {
  union { float f; unsigned int u; } v; v.f = f;
  unsigned int u = v.u;
  return (unsigned short)((u + 0x7FFFu + ((u >> 16) & 1u)) >> 16);
}
static __device__ __forceinline__ __bf16 us2bf(unsigned short s) {
  union { unsigned short s; __bf16 b; } v; v.s = s; return v.b;
}

// ---------------------------------------------------------------------------
// Kernel 1: f32 -> bf16 weight conversion
// ---------------------------------------------------------------------------
__global__ void f32_to_bf16_kernel(const float* __restrict__ s,
                                   unsigned short* __restrict__ d, int n) {
  int i = blockIdx.x * 256 + threadIdx.x;
  if (i < n) d[i] = f2bf(s[i]);
}

// ---------------------------------------------------------------------------
// Kernel 2: 3-NN over n_g=2048 group points (coord2 tile fully in LDS) +
//           inverse-distance weights.  One thread per query point.
// ---------------------------------------------------------------------------
__global__ void __launch_bounds__(256)
knn3_kernel(const float* __restrict__ c1, const float* __restrict__ c2,
            int* __restrict__ idx, float* __restrict__ wgt) {
  __shared__ float sx[2048], sy[2048], sz[2048], sn[2048];
  const int b = blockIdx.y;
  const int p = blockIdx.x * 256 + threadIdx.x;
  const float* c2b = c2 + (size_t)b * 3 * 2048;
  for (int i = threadIdx.x; i < 2048; i += 256) {
    float x = c2b[i], y = c2b[2048 + i], z = c2b[4096 + i];
    sx[i] = x; sy[i] = y; sz[i] = z; sn[i] = x * x + y * y + z * z;
  }
  __syncthreads();
  const float* c1b = c1 + (size_t)b * 3 * 8192;
  float x = c1b[p], y = c1b[8192 + p], z = c1b[16384 + p];
  float pn = x * x + y * y + z * z;
  float d0 = 3.4e38f, d1 = 3.4e38f, d2 = 3.4e38f;
  int i0 = 0, i1 = 0, i2 = 0;
  for (int g = 0; g < 2048; ++g) {
    float d = pn + sn[g] - 2.0f * (x * sx[g] + y * sy[g] + z * sz[g]);
    if (d < d0)      { d2 = d1; i2 = i1; d1 = d0; i1 = i0; d0 = d; i0 = g; }
    else if (d < d1) { d2 = d1; i2 = i1; d1 = d;  i1 = g; }
    else if (d < d2) { d2 = d;  i2 = g; }
  }
  float r0 = 1.0f / (d0 + 1e-8f), r1 = 1.0f / (d1 + 1e-8f), r2 = 1.0f / (d2 + 1e-8f);
  float rs = 1.0f / (r0 + r1 + r2);
  size_t o = ((size_t)b * 8192 + p) * 3;
  idx[o] = i0; idx[o + 1] = i1; idx[o + 2] = i2;
  wgt[o] = r0 * rs; wgt[o + 1] = r1 * rs; wgt[o + 2] = r2 * rs;
}

// ---------------------------------------------------------------------------
// Kernel 3: build fused feat matrix [B, 384, n_p] in bf16:
//           rows 0..127  = feature1, rows 128..383 = IDW-interp of feature2
// ---------------------------------------------------------------------------
__global__ void __launch_bounds__(256)
build_feat_kernel(const float* __restrict__ f1, const float* __restrict__ f2,
                  const int* __restrict__ idx, const float* __restrict__ wgt,
                  unsigned short* __restrict__ featB) {
  const int b = blockIdx.y;
  const int p = blockIdx.x * 256 + threadIdx.x;
  size_t o = ((size_t)b * 8192 + p) * 3;
  const int g0 = idx[o], g1 = idx[o + 1], g2 = idx[o + 2];
  const float w0 = wgt[o], w1 = wgt[o + 1], w2 = wgt[o + 2];
  const float* f1b = f1 + (size_t)b * 128 * 8192;
  unsigned short* fo = featB + (size_t)b * 384 * 8192;
  for (int c = 0; c < 128; ++c)
    fo[(size_t)c * 8192 + p] = f2bf(f1b[(size_t)c * 8192 + p]);
  const float* f2b = f2 + (size_t)b * 256 * 2048;
  for (int c = 0; c < 256; ++c) {
    const float* row = f2b + (size_t)c * 2048;
    float v = w0 * row[g0] + w1 * row[g1] + w2 * row[g2];
    fo[(size_t)(128 + c) * 8192 + p] = f2bf(v);
  }
}

// ---------------------------------------------------------------------------
// Kernel 4: WMMA bf16 GEMM   Y[b,m,n] = sum_k W[m,k] * X[b,k,n] + bias[m]
//           N fixed = 8192.  Block = 256 threads = 8 waves; block tile
//           M=128 x N=64; each wave: 16(M) x 64(N) via 4 accumulators.
// ---------------------------------------------------------------------------
__global__ void __launch_bounds__(256)
gemm_bf16_kernel(const unsigned short* __restrict__ X,  // [B,K,8192] bf16
                 const unsigned short* __restrict__ W,  // [M,K] bf16
                 const float* __restrict__ bias,        // [M]
                 float* __restrict__ Y,                 // [B,M,8192] f32
                 int M, int K) {
  __shared__ unsigned short s_x[64 * 40];   // [n][k], stride 40
  __shared__ unsigned short s_w[128 * 40];  // [m][k], stride 40
  const int tid  = threadIdx.x;
  const int lane = tid & 31;
  const int wave = tid >> 5;
  const int l15  = lane & 15;
  const int n0   = blockIdx.x * 64;
  const int m0   = blockIdx.y * 128;
  const int b    = blockIdx.z;
  const size_t xbase = (size_t)b * K * 8192;

  v8f acc[4] = {};

  const int mloc = wave * 16 + l15;

  for (int k0 = 0; k0 < K; k0 += 32) {
    // ---- stage X tile [32 k][64 n] -> LDS [n][k] ----
    {
      const int kk = tid >> 3;          // 0..31
      const int cc = (tid & 7) * 8;     // 0..56
      union { uint4 v; unsigned short e[8]; } u;
      u.v = *reinterpret_cast<const uint4*>(X + xbase + (size_t)(k0 + kk) * 8192 + n0 + cc);
      #pragma unroll
      for (int ii = 0; ii < 8; ++ii)
        s_x[(cc + ii) * 40 + kk] = u.e[ii];
    }
    // ---- stage W tile [128 m][32 k] -> LDS [m][k] ----
    #pragma unroll
    for (int it = 0; it < 2; ++it) {
      const int e    = tid + it * 256;  // 0..511
      const int m    = e >> 2;          // 0..127
      const int part = (e & 3) * 8;     // 0,8,16,24
      uint4 v = *reinterpret_cast<const uint4*>(W + (size_t)(m0 + m) * K + k0 + part);
      *reinterpret_cast<uint4*>(&s_w[m * 40 + part]) = v;
    }
    if (k0 + 32 < K)  // gfx1250 global_prefetch_b8 on next K tile
      __builtin_prefetch(X + xbase + (size_t)(k0 + 32 + (tid >> 3)) * 8192 + n0 + (tid & 7) * 8, 0, 1);
    __syncthreads();

    // ---- A fragment: 16x32 bf16, row m = lane%16.
    //      lanes 0-15: VGPR0-3 K=0..7,  VGPR4-7 K=16..23
    //      lanes16-31: VGPR0-3 K=8..15, VGPR4-7 K=24..31
    v16bf a{};
    {
      const int kb0 = (lane < 16) ? 0 : 8;
      const int kb1 = (lane < 16) ? 16 : 24;
      const unsigned short* wr = &s_w[mloc * 40];
      #pragma unroll
      for (int j = 0; j < 4; ++j) {
        a[2 * j]         = us2bf(wr[kb0 + 2 * j]);
        a[2 * j + 1]     = us2bf(wr[kb0 + 2 * j + 1]);
        a[8 + 2 * j]     = us2bf(wr[kb1 + 2 * j]);
        a[8 + 2 * j + 1] = us2bf(wr[kb1 + 2 * j + 1]);
      }
    }
    // ---- B fragments: 32x16 bf16, col n = lane%16,
    //      lanes 0-15 hold K=0..7,16..23, lanes 16-31 hold K=8..15,24..31
    const int kb = (lane < 16) ? 0 : 8;
    #pragma unroll
    for (int ns = 0; ns < 4; ++ns) {
      const unsigned short* xr = &s_x[(ns * 16 + l15) * 40 + kb];
      v16bf bf{};
      #pragma unroll
      for (int j = 0; j < 16; ++j) bf[j] = us2bf(xr[(j < 8) ? j : j + 8]);
      acc[ns] = __builtin_amdgcn_wmma_f32_16x16x32_bf16(
          false, a, false, bf, (short)0, acc[ns], false, false);
    }
    __syncthreads();
  }

  // ---- epilogue: 16x16 f32 C/D layout: VGPR j -> M = j (lanes 0-15) / j+8
  #pragma unroll
  for (int ns = 0; ns < 4; ++ns) {
    #pragma unroll
    for (int j = 0; j < 8; ++j) {
      const int mrow = m0 + wave * 16 + ((lane < 16) ? j : j + 8);
      const int ncol = n0 + ns * 16 + l15;
      Y[(size_t)(b * M + mrow) * 8192 + ncol] = acc[ns][j] + bias[mrow];
    }
  }
}

// ---------------------------------------------------------------------------
// Kernel 5: per-channel mean/var over (B, n_p)  (train-mode BatchNorm stats)
// ---------------------------------------------------------------------------
__global__ void __launch_bounds__(256)
bn_stats_kernel(const float* __restrict__ Xp, int C, float* __restrict__ stat) {
  __shared__ float ssum[256], ssq[256];
  const int c = blockIdx.x;
  float s = 0.0f, q = 0.0f;
  for (int i = threadIdx.x; i < 16 * 8192; i += 256) {
    const int b = i >> 13, n = i & 8191;
    float v = Xp[((size_t)b * C + c) * 8192 + n];
    s += v; q += v * v;
  }
  ssum[threadIdx.x] = s; ssq[threadIdx.x] = q;
  __syncthreads();
  for (int st = 128; st > 0; st >>= 1) {
    if (threadIdx.x < st) {
      ssum[threadIdx.x] += ssum[threadIdx.x + st];
      ssq[threadIdx.x]  += ssq[threadIdx.x + st];
    }
    __syncthreads();
  }
  if (threadIdx.x == 0) {
    const float inv = 1.0f / (16.0f * 8192.0f);
    float mean = ssum[0] * inv;
    float var  = fmaxf(ssq[0] * inv - mean * mean, 0.0f);
    stat[c * 2] = mean; stat[c * 2 + 1] = var;
  }
}

// ---------------------------------------------------------------------------
// Kernel 6a/6b: apply BN + ReLU; bf16 output (feeds GEMM2) or f32 (final)
// ---------------------------------------------------------------------------
__global__ void __launch_bounds__(256)
bn_apply_bf16_kernel(const float* __restrict__ Xp, const float* __restrict__ stat,
                     const float* __restrict__ gamma, const float* __restrict__ beta,
                     unsigned short* __restrict__ Yb, int C) {
  const int n = blockIdx.x * 256 + threadIdx.x;
  const int c = blockIdx.y, b = blockIdx.z;
  const float mean = stat[c * 2], var = stat[c * 2 + 1];
  const float g  = gamma[c] * rsqrtf(var + 1e-5f);
  const float bt = beta[c] - mean * g;
  float v = Xp[((size_t)b * C + c) * 8192 + n];
  Yb[((size_t)b * C + c) * 8192 + n] = f2bf(fmaxf(v * g + bt, 0.0f));
}

__global__ void __launch_bounds__(256)
bn_apply_f32_kernel(const float* __restrict__ Xp, const float* __restrict__ stat,
                    const float* __restrict__ gamma, const float* __restrict__ beta,
                    float* __restrict__ Y, int C) {
  const int n = blockIdx.x * 256 + threadIdx.x;
  const int c = blockIdx.y, b = blockIdx.z;
  const float mean = stat[c * 2], var = stat[c * 2 + 1];
  const float g  = gamma[c] * rsqrtf(var + 1e-5f);
  const float bt = beta[c] - mean * g;
  float v = Xp[((size_t)b * C + c) * 8192 + n];
  Y[((size_t)b * C + c) * 8192 + n] = fmaxf(v * g + bt, 0.0f);
}

// ---------------------------------------------------------------------------
// Launch
// ---------------------------------------------------------------------------
extern "C" void kernel_launch(void* const* d_in, const int* in_sizes, int n_in,
                              void* d_out, int out_size, void* d_ws, size_t ws_size,
                              hipStream_t stream) {
  (void)in_sizes; (void)n_in; (void)out_size; (void)ws_size;
  const float* feature1 = (const float*)d_in[0];
  const float* coord1   = (const float*)d_in[1];
  const float* feature2 = (const float*)d_in[2];
  const float* coord2   = (const float*)d_in[3];
  const float* W1     = (const float*)d_in[4];
  const float* b1     = (const float*)d_in[5];
  const float* gamma1 = (const float*)d_in[6];
  const float* beta1  = (const float*)d_in[7];
  const float* W2     = (const float*)d_in[8];
  const float* b2     = (const float*)d_in[9];
  const float* gamma2 = (const float*)d_in[10];
  const float* beta2  = (const float*)d_in[11];

  char* ws = (char*)d_ws;
  unsigned short* featB = (unsigned short*)(ws + 0);          // 100,663,296 B
  float*          h_pre = (float*)(ws + 100663296);           // 134,217,728 B
  unsigned short* hB    = (unsigned short*)(ws + 234881024);  //  67,108,864 B
  unsigned short* W1b   = (unsigned short*)(ws + 301989888);  //     196,608 B
  unsigned short* W2b   = (unsigned short*)(ws + 302186496);  //      65,536 B
  int*            knn_i = (int*)(ws + 302252032);             //   1,572,864 B
  float*          knn_w = (float*)(ws + 303824896);           //   1,572,864 B
  float*          stat1 = (float*)(ws + 305397760);           //       2,048 B
  float*          stat2 = (float*)(ws + 305399808);           //       1,024 B
  float*          o_pre = (float*)(ws + 0);                   // aliases featB (dead after GEMM1)

  f32_to_bf16_kernel<<<384, 256, 0, stream>>>(W1, W1b, 256 * 384);
  f32_to_bf16_kernel<<<128, 256, 0, stream>>>(W2, W2b, 128 * 256);

  knn3_kernel<<<dim3(32, 16), 256, 0, stream>>>(coord1, coord2, knn_i, knn_w);
  build_feat_kernel<<<dim3(32, 16), 256, 0, stream>>>(feature1, feature2, knn_i, knn_w, featB);

  gemm_bf16_kernel<<<dim3(128, 2, 16), 256, 0, stream>>>(featB, W1b, b1, h_pre, 256, 384);
  bn_stats_kernel<<<256, 256, 0, stream>>>(h_pre, 256, stat1);
  bn_apply_bf16_kernel<<<dim3(32, 256, 16), 256, 0, stream>>>(h_pre, stat1, gamma1, beta1, hB, 256);

  gemm_bf16_kernel<<<dim3(128, 1, 16), 256, 0, stream>>>(hB, W2b, b2, o_pre, 128, 256);
  bn_stats_kernel<<<128, 256, 0, stream>>>(o_pre, 128, stat2);
  bn_apply_f32_kernel<<<dim3(32, 128, 16), 256, 0, stream>>>(o_pre, stat2, gamma2, beta2,
                                                             (float*)d_out, 128);
}
